// _Network_50646254354551
// MI455X (gfx1250) — compile-verified, matrix operand order checked by reading, not run on hardware
//
#include <hip/hip_runtime.h>
#include <hip/hip_bf16.h>

// CDNA5 / gfx1250 fused kernel: shared-MLP embedding + 5-head neighbor
// attention + tail MLP. All dense layers on v_wmma_f32_16x16x32_bf16
// (bf16 in, f32 accumulate). Weights converted to bf16 ONCE per block into
// LDS (transposed, conflict-free stride); every WMMA fragment is then two
// ds_load_b128. Softmax-attention runs on the VALU with per-wave LDS staging.

typedef __attribute__((ext_vector_type(16))) __bf16 v16bf;
typedef __attribute__((ext_vector_type(8)))  float  v8f;

#define WAVES   4
#define TPB     (WAVES * 32)
#define TILE_B  16
#define EDIM    32
#define NHEAD   5
#define PDIM    8
#define NNEI    4
#define FIN     20

#define WSTRIDE 40                 // bf16 row stride: 80 B, 16B-aligned, bank-conflict-free
#define WMAT    (EDIM * WSTRIDE)   // one staged 32x32 matrix (1280 bf16)
#define NMAT    19                 // emb1, emb2, (q,k,v)x5, tail1, tail2

// weight-matrix slot indices in LDS
#define W_EMB1  0
#define W_EMB2  1
#define W_Q(h)  (2 + 3 * (h))
#define W_K(h)  (3 + 3 * (h))
#define W_V(h)  (4 + 3 * (h))
#define W_T1    17
#define W_T2    18

// Per-wave scratch. stage (phases 1/3) and qbuf (phase 2) never live at the
// same time -> union. kbuf/vbuf strides padded against bank conflicts.
struct alignas(16) WaveScratch {
  union {
    float  qbuf [TILE_B * 33];
    __bf16 stage[TILE_B * WSTRIDE];
  };
  __bf16 kbuf[64 * 34];
  __bf16 vbuf[64 * 34];
};

static __device__ __forceinline__ void waitds() {
  // DS ops within a wave are serviced in order; this both drains the counter
  // and stops the compiler reordering LDS traffic across it.
  asm volatile("s_wait_dscnt 0" ::: "memory");
}

static __device__ __forceinline__ v8f vzero() {
  v8f z;
#pragma unroll
  for (int i = 0; i < 8; ++i) z[i] = 0.f;
  return z;
}

static __device__ __forceinline__ v8f wmma_bf16(v16bf a, v16bf b, v8f c) {
  // (neg_a, A, neg_b, B, c_mod, C, reuse_a, reuse_b)
  return __builtin_amdgcn_wmma_f32_16x16x32_bf16(false, a, false, b,
                                                 (short)0, c, false, false);
}

// Generic 16x32 fragment load from a stride-40 bf16 LDS tile.
// Fragment lane mapping (A and B are mirrors): lane holds row `row` and the
// two contiguous 8-element K-runs [8*ksel, +7] and [16+8*ksel, +7]
// -> two b128 LDS loads.
static __device__ __forceinline__ v16bf load_frag(const __bf16* base,
                                                  int row, int ksel) {
  union { uint4 q[2]; v16bf v; } u;
  u.q[0] = *(const uint4*)(base + row * WSTRIDE + 8 * ksel);
  u.q[1] = *(const uint4*)(base + row * WSTRIDE + 16 + 8 * ksel);
  return u.v;
}

// A-fragment built directly from global f32 features (row stride FIN=20,
// K zero-padded to 32). rowmax clamps partial tiles.
static __device__ __forceinline__ v16bf load_a_feat(const float* __restrict__ x,
                                                    long row0, long rowmax,
                                                    int lane) {
  const int m = lane & 15, ksel = lane >> 4;
  long r = row0 + m;
  if (r >= rowmax) r = rowmax - 1;
  const float* row = x + r * FIN;
  v16bf a;
#pragma unroll
  for (int j = 0; j < 8; ++j) {
    const int k = ((j < 4) ? (2 * j) : (16 + 2 * (j - 4))) + 8 * ksel;
    const float lo = (k     < FIN) ? row[k]     : 0.f;
    const float hi = (k + 1 < FIN) ? row[k + 1] : 0.f;
    a[2 * j]     = (__bf16)lo;
    a[2 * j + 1] = (__bf16)hi;
  }
  return a;
}

// Cooperative block-wide staging: global f32 w[k*ld + n] -> LDS bf16,
// TRANSPOSED to dst[n*WSTRIDE + k], zero-padded past kmax/nmax.
// Consecutive threads read consecutive n -> coalesced global loads.
static __device__ __forceinline__ void stage_weight(__bf16* dst,
                                                    const float* __restrict__ src,
                                                    int ld, int kmax, int nmax,
                                                    int tid) {
  for (int idx = tid; idx < EDIM * EDIM; idx += TPB) {
    const int k = idx >> 5, n = idx & 31;
    const float v = (k < kmax && n < nmax) ? src[k * ld + n] : 0.f;
    dst[n * WSTRIDE + k] = (__bf16)v;
  }
}

static __device__ __forceinline__ void stage_bias(float* dst,
                                                  const float* __restrict__ src,
                                                  int cnt, int tid) {
  if (tid < EDIM) dst[tid] = (tid < cnt) ? src[tid] : 0.f;
}

// Accumulator pair (N-tiles 0/1) -> destination, + bias + ReLU.
// Acc layout: lane holds col n = (lane&15)+16*tile, rows m = r + 8*(lane>>4).
static __device__ __forceinline__ void acc_to_bf16(__bf16* buf, int stride,
                                                   const v8f& c0, const v8f& c1,
                                                   const float* bias, int lane) {
  const int n0 = lane & 15, mh = (lane >> 4) * 8;
  const float b0 = bias[n0], b1 = bias[n0 + 16];
#pragma unroll
  for (int r = 0; r < 8; ++r) {
    buf[(mh + r) * stride + n0]      = (__bf16)fmaxf(c0[r] + b0, 0.f);
    buf[(mh + r) * stride + n0 + 16] = (__bf16)fmaxf(c1[r] + b1, 0.f);
  }
}

static __device__ __forceinline__ void acc_to_qbuf(float* buf,
                                                   const v8f& c0, const v8f& c1,
                                                   const float* bias, int lane) {
  const int n0 = lane & 15, mh = (lane >> 4) * 8;
  const float b0 = bias[n0], b1 = bias[n0 + 16];
#pragma unroll
  for (int r = 0; r < 8; ++r) {
    buf[(mh + r) * 33 + n0]      = fmaxf(c0[r] + b0, 0.f);
    buf[(mh + r) * 33 + n0 + 16] = fmaxf(c1[r] + b1, 0.f);
  }
}

__global__ __launch_bounds__(TPB) void fused_attention_qnet(
    const float* __restrict__ agent_feat, const float* __restrict__ neigh_feat,
    const float* __restrict__ emb_w1, const float* __restrict__ emb_b1,
    const float* __restrict__ emb_w2, const float* __restrict__ emb_b2,
    const float* __restrict__ wq, const float* __restrict__ bq,
    const float* __restrict__ wk, const float* __restrict__ bk,
    const float* __restrict__ wv, const float* __restrict__ bv,
    const float* __restrict__ tail_w1, const float* __restrict__ tail_b1,
    const float* __restrict__ tail_w2, const float* __restrict__ tail_b2,
    float* __restrict__ out, int Btot)
{
  __shared__ __bf16 wlds[NMAT * WMAT];     // 48.6 KB: all weights, bf16, transposed
  __shared__ float  blds[NMAT * EDIM];     //  2.4 KB: all biases
  __shared__ WaveScratch wsc[WAVES];       // 43.3 KB: per-wave activation scratch

  const int tid  = threadIdx.x;
  const int wave = tid >> 5;
  const int lane = tid & 31;

  // ---- block-wide one-time weight/bias staging (f32 -> bf16, transpose) ----
  stage_weight(wlds + W_EMB1 * WMAT, emb_w1, EDIM, FIN,  EDIM, tid);
  stage_weight(wlds + W_EMB2 * WMAT, emb_w2, EDIM, EDIM, EDIM, tid);
#pragma unroll
  for (int h = 0; h < NHEAD; ++h) {
    stage_weight(wlds + W_Q(h) * WMAT, wq + (long)h * EDIM * EDIM, EDIM, EDIM, EDIM, tid);
    stage_weight(wlds + W_K(h) * WMAT, wk + (long)h * EDIM * EDIM, EDIM, EDIM, EDIM, tid);
    stage_weight(wlds + W_V(h) * WMAT, wv + (long)h * EDIM * EDIM, EDIM, EDIM, EDIM, tid);
  }
  stage_weight(wlds + W_T1 * WMAT, tail_w1, EDIM, EDIM, EDIM, tid);
  stage_weight(wlds + W_T2 * WMAT, tail_w2, PDIM, EDIM, PDIM, tid);

  stage_bias(blds + W_EMB1 * EDIM, emb_b1, EDIM, tid);
  stage_bias(blds + W_EMB2 * EDIM, emb_b2, EDIM, tid);
#pragma unroll
  for (int h = 0; h < NHEAD; ++h) {
    stage_bias(blds + W_Q(h) * EDIM, bq + h * EDIM, EDIM, tid);
    stage_bias(blds + W_K(h) * EDIM, bk + h * EDIM, EDIM, tid);
    stage_bias(blds + W_V(h) * EDIM, bv + h * EDIM, EDIM, tid);
  }
  stage_bias(blds + W_T1 * EDIM, tail_b1, EDIM, tid);
  stage_bias(blds + W_T2 * EDIM, tail_b2, PDIM, tid);
  __syncthreads();

  const long b0 = ((long)blockIdx.x * WAVES + wave) * TILE_B;
  if (b0 >= Btot) return;                      // wave-uniform exit (after barrier)
  WaveScratch& ws = wsc[wave];

  const int frow  = lane & 15;                 // fragment row within a 16-tile
  const int fksel = lane >> 4;                 // which K-interleave half

  // ---------------- Phase 1: shared embedding MLP (WMMA) ----------------
  const v16bf bw10 = load_frag(wlds + W_EMB1 * WMAT, frow,      fksel);
  const v16bf bw11 = load_frag(wlds + W_EMB1 * WMAT, frow + 16, fksel);
  const v16bf bw20 = load_frag(wlds + W_EMB2 * WMAT, frow,      fksel);
  const v16bf bw21 = load_frag(wlds + W_EMB2 * WMAT, frow + 16, fksel);

  v16bf embA[5];   // A-fragments: [0]=agent tile, [1..4]=neighbor tiles
#pragma unroll
  for (int t = 0; t < 5; ++t) {
    v16bf a = (t == 0)
        ? load_a_feat(agent_feat, b0, (long)Btot, lane)
        : load_a_feat(neigh_feat, b0 * NNEI + (long)(t - 1) * 16,
                      (long)Btot * NNEI, lane);
    v8f c0 = vzero(), c1 = vzero();
    c0 = wmma_bf16(a, bw10, c0);
    c1 = wmma_bf16(a, bw11, c1);
    acc_to_bf16(ws.stage, WSTRIDE, c0, c1, blds + W_EMB1 * EDIM, lane);
    waitds();
    v16bf h1 = load_frag(ws.stage, frow, fksel);
    waitds();
    c0 = vzero(); c1 = vzero();
    c0 = wmma_bf16(h1, bw20, c0);
    c1 = wmma_bf16(h1, bw21, c1);
    acc_to_bf16(ws.stage, WSTRIDE, c0, c1, blds + W_EMB2 * EDIM, lane);
    waitds();
    embA[t] = load_frag(ws.stage, frow, fksel);
    waitds();
  }

  // ---------------- Phase 2: heads (q/k/v on WMMA, attention on VALU) ----
  const int a_id = lane & 15;          // local agent owned by this lane
  const int e0   = (lane >> 4) * 16;   // this lane's half of the E dim
  float hsum[16];
#pragma unroll
  for (int j = 0; j < 16; ++j) hsum[j] = 0.f;

  for (int h = 0; h < NHEAD; ++h) {
    {   // q = relu(agent_emb @ wq + bq) -> qbuf (f32)
      const __bf16* W = wlds + W_Q(h) * WMAT;
      v16bf f0 = load_frag(W, frow,      fksel);
      v16bf f1 = load_frag(W, frow + 16, fksel);
      v8f c0 = vzero(), c1 = vzero();
      c0 = wmma_bf16(embA[0], f0, c0);
      c1 = wmma_bf16(embA[0], f1, c1);
      acc_to_qbuf(ws.qbuf, c0, c1, blds + W_Q(h) * EDIM, lane);
    }
    {   // k = relu(neigh_emb @ wk + bk) -> kbuf (bf16)
      const __bf16* W = wlds + W_K(h) * WMAT;
      v16bf f0 = load_frag(W, frow,      fksel);
      v16bf f1 = load_frag(W, frow + 16, fksel);
#pragma unroll
      for (int t = 0; t < 4; ++t) {
        v8f c0 = vzero(), c1 = vzero();
        c0 = wmma_bf16(embA[1 + t], f0, c0);
        c1 = wmma_bf16(embA[1 + t], f1, c1);
        acc_to_bf16(ws.kbuf + t * 16 * 34, 34, c0, c1, blds + W_K(h) * EDIM, lane);
      }
    }
    {   // v = relu(neigh_emb @ wv + bv) -> vbuf (bf16)
      const __bf16* W = wlds + W_V(h) * WMAT;
      v16bf f0 = load_frag(W, frow,      fksel);
      v16bf f1 = load_frag(W, frow + 16, fksel);
#pragma unroll
      for (int t = 0; t < 4; ++t) {
        v8f c0 = vzero(), c1 = vzero();
        c0 = wmma_bf16(embA[1 + t], f0, c0);
        c1 = wmma_bf16(embA[1 + t], f1, c1);
        acc_to_bf16(ws.vbuf + t * 16 * 34, 34, c0, c1, blds + W_V(h) * EDIM, lane);
      }
    }
    waitds();

    // scores + softmax + weighted sum; lane owns (agent a_id, e-half e0).
    float qv[16];
#pragma unroll
    for (int j = 0; j < 16; ++j) qv[j] = ws.qbuf[a_id * 33 + e0 + j];

    float sc[NNEI];
#pragma unroll
    for (int n = 0; n < NNEI; ++n) {
      const __bf16* kr = ws.kbuf + (a_id * NNEI + n) * 34 + e0;
      float s = 0.f;
#pragma unroll
      for (int j = 0; j < 16; ++j) s += qv[j] * (float)kr[j];
      s += __shfl_xor(s, 16, 32);   // combine the two e-halves
      sc[n] = s;
    }
    const float mx = fmaxf(fmaxf(sc[0], sc[1]), fmaxf(sc[2], sc[3]));
    float ex[NNEI], den = 0.f;
#pragma unroll
    for (int n = 0; n < NNEI; ++n) { ex[n] = __expf(sc[n] - mx); den += ex[n]; }
    const float rden = 1.f / den;
#pragma unroll
    for (int n = 0; n < NNEI; ++n) {
      const float att = ex[n] * rden;
      const __bf16* vr = ws.vbuf + (a_id * NNEI + n) * 34 + e0;
#pragma unroll
      for (int j = 0; j < 16; ++j) hsum[j] += att * (float)vr[j];
    }
    waitds();
  }

  // ---------------- Phase 3: head-mean + tail MLP (WMMA) -----------------
#pragma unroll
  for (int j = 0; j < 16; ++j)
    ws.stage[a_id * WSTRIDE + e0 + j] = (__bf16)(hsum[j] * (1.f / NHEAD));
  waitds();
  v16bf xa = load_frag(ws.stage, frow, fksel);
  waitds();
  {
    v16bf f0 = load_frag(wlds + W_T1 * WMAT, frow,      fksel);
    v16bf f1 = load_frag(wlds + W_T1 * WMAT, frow + 16, fksel);
    v8f c0 = vzero(), c1 = vzero();
    c0 = wmma_bf16(xa, f0, c0);
    c1 = wmma_bf16(xa, f1, c1);
    acc_to_bf16(ws.stage, WSTRIDE, c0, c1, blds + W_T1 * EDIM, lane);
  }
  waitds();
  v16bf ha = load_frag(ws.stage, frow, fksel);
  waitds();
  {
    v16bf f2 = load_frag(wlds + W_T2 * WMAT, frow, fksel);   // N padded 8->16
    v8f o = vzero();
    o = wmma_bf16(ha, f2, o);
    const int n = lane & 15;
    if (n < PDIM) {
      const int mh = (lane >> 4) * 8;
      const float bo = blds[W_T2 * EDIM + n];
#pragma unroll
      for (int r = 0; r < 8; ++r) {
        const long row = b0 + mh + r;
        if (row < Btot) out[row * PDIM + n] = o[r] + bo;
      }
    }
  }
}

extern "C" void kernel_launch(void* const* d_in, const int* in_sizes, int n_in,
                              void* d_out, int out_size, void* d_ws, size_t ws_size,
                              hipStream_t stream) {
  (void)n_in; (void)out_size; (void)d_ws; (void)ws_size;
  const float* agent_feat = (const float*)d_in[0];
  const float* neigh_feat = (const float*)d_in[1];
  const float* emb_w1     = (const float*)d_in[2];
  const float* emb_b1     = (const float*)d_in[3];
  const float* emb_w2     = (const float*)d_in[4];
  const float* emb_b2     = (const float*)d_in[5];
  const float* wq         = (const float*)d_in[6];
  const float* bq         = (const float*)d_in[7];
  const float* wk         = (const float*)d_in[8];
  const float* bk         = (const float*)d_in[9];
  const float* wv         = (const float*)d_in[10];
  const float* bv         = (const float*)d_in[11];
  const float* tail_w1    = (const float*)d_in[12];
  const float* tail_b1    = (const float*)d_in[13];
  const float* tail_w2    = (const float*)d_in[14];
  const float* tail_b2    = (const float*)d_in[15];
  float* out = (float*)d_out;

  const int Btot = in_sizes[0] / FIN;            // 262144
  const int per_block = WAVES * TILE_B;          // 64 elements / block
  const int grid = (Btot + per_block - 1) / per_block;

  fused_attention_qnet<<<grid, TPB, 0, stream>>>(
      agent_feat, neigh_feat, emb_w1, emb_b1, emb_w2, emb_b2,
      wq, bq, wk, bk, wv, bv, tail_w1, tail_b1, tail_w2, tail_b2,
      out, Btot);
}
